// SparseAttention_61718680044329
// MI455X (gfx1250) — matrix operand order repaired and checked by reading, not compiled, not run on hardware
//
#include <hip/hip_runtime.h>
#include <stdint.h>

#define DEV __device__ __forceinline__

typedef __attribute__((ext_vector_type(16))) __bf16        v16bf;
typedef __attribute__((ext_vector_type(8)))  float         v8f;
typedef __attribute__((ext_vector_type(4)))  unsigned int  u32x4;

union Frag16 { v16bf v; unsigned short u[16]; u32x4 q[2]; };

static constexpr int BATCH = 4;
static constexpr int NTOK  = 8192;
static constexpr int DIM   = 512;
static constexpr int HEADS = 8;
static constexpr int HD    = 64;     // head dim
static constexpr int WS    = 256;    // window size
static constexpr int NW    = 32;     // windows per batch
static constexpr int G     = 32;     // global tokens
static constexpr int KKEEP = 179;    // int(256 * (1 - 0.3))
static constexpr float SCALE = 0.044194173824159216f;  // 512^-0.5

DEV unsigned short f2bf(float f) {
  unsigned u = __float_as_uint(f);
  u += 0x7FFFu + ((u >> 16) & 1u);   // round-to-nearest-even
  return (unsigned short)(u >> 16);
}
DEV float bf2f(unsigned short s) { return __uint_as_float(((unsigned)s) << 16); }
DEV unsigned keyOf(float f) {        // order-preserving float -> uint map
  unsigned u = __float_as_uint(f);
  return (u & 0x80000000u) ? ~u : (u | 0x80000000u);
}
DEV unsigned lds_off(const void* p) {
  return (unsigned)(size_t)(__attribute__((address_space(3))) const void*)p;
}

// CDNA5 async global->LDS copy (ASYNCcnt-tracked, LDS written by memory system).
DEV void async_copy_b128(const void* g, void* lds) {
  asm volatile("global_load_async_to_lds_b128 %0, %1, off"
               :: "v"(lds_off(lds)), "v"(g) : "memory");
}
DEV void wait_async0() { asm volatile("s_wait_asynccnt 0x0" ::: "memory"); }
DEV void wait_async1() { asm volatile("s_wait_asynccnt 0x1" ::: "memory"); }

// ---------------- CDNA5 Tensor Data Mover (TDM) 2D tile load, bf16 elements
#if defined(__has_builtin)
#if __has_builtin(__builtin_amdgcn_tensor_load_to_lds)
#define HAVE_TDM 1
#endif
#endif

#ifdef HAVE_TDM
typedef __attribute__((ext_vector_type(4))) unsigned int tdm_v4u;
typedef __attribute__((ext_vector_type(8))) int          tdm_v8i;
typedef __attribute__((ext_vector_type(4))) int          tdm_v4i;

// D# per CDNA5 ISA: 2D tile of 2-byte elements.
// This toolchain exposes the 6-arg builtin:
//   (uint32x4 g0, int32x8 g1, int32x4 g2, int32x4 g3, int32x8 g4, i32 cpol)
DEV void tdm_load_2d_bf16(const void* gaddr, unsigned lds_byte_off,
                          unsigned tile_w, unsigned tile_h, unsigned row_stride) {
  uint64_t ga = (uint64_t)gaddr;
  tdm_v4u g0;
  g0[0] = 1u;                                            // count=1, user mode
  g0[1] = lds_byte_off;                                  // lds_addr
  g0[2] = (unsigned)ga;                                  // global_addr[31:0]
  g0[3] = (unsigned)((ga >> 32) & 0x01FFFFFFu) | (2u << 30);  // [56:32] | type=2
  tdm_v8i g1;
  g1[0] = (int)(1u << 16);                               // wg_mask=0, data_size=2B
  g1[1] = (int)((row_stride & 0xFFFFu) << 16);           // tensor_dim0[15:0]
  g1[2] = (int)(((row_stride >> 16) & 0xFFFFu) | ((tile_h & 0xFFFFu) << 16));
  g1[3] = (int)(((tile_h >> 16) & 0xFFFFu) | ((tile_w & 0xFFFFu) << 16));  // tile_dim0
  g1[4] = (int)(tile_h & 0xFFFFu);                       // tile_dim1 (tile_dim2=0)
  g1[5] = (int)row_stride;                               // tensor_dim0_stride lo
  g1[6] = 0;
  g1[7] = 0;
  tdm_v4i z4 = {0, 0, 0, 0};
  tdm_v8i z8 = {0, 0, 0, 0, 0, 0, 0, 0};
  __builtin_amdgcn_tensor_load_to_lds(g0, g1, z4, z4, z8, 0);
}
#endif

DEV void wait_tensor0() {
#if defined(__has_builtin)
#if __has_builtin(__builtin_amdgcn_s_wait_tensorcnt)
  __builtin_amdgcn_s_wait_tensorcnt(0);
  return;
#endif
#endif
  asm volatile("s_wait_tensorcnt 0x0" ::: "memory");
}

// ---------------------------------------------------------------- converts
__global__ void cvt_f32_bf16(const float* __restrict__ in,
                             unsigned short* __restrict__ out, int n) {
  for (int i = blockIdx.x * blockDim.x + threadIdx.x; i < n;
       i += gridDim.x * blockDim.x)
    out[i] = f2bf(in[i]);
}

// ---------------------------------------------------------------- bf16 GEMM
// C[M,N] = A[M,K] @ B[K,N]; bf16 in, f32 accumulate. Block tile 64x64, BK=32.
// Double-buffered: A tiles arrive via async global->LDS (next tile in flight
// during compute, drained with partial s_wait_asynccnt 0x1); B next-chunk kept
// in VGPRs across the barrier.
template <bool OUT_BF16>
__global__ __launch_bounds__(256) void gemm_bf16(
    const unsigned short* __restrict__ A, const unsigned short* __restrict__ B,
    float* __restrict__ Cf, unsigned short* __restrict__ Cb,
    const float* __restrict__ bias, int M, int N, int K) {
  __shared__ unsigned short Al[2][64 * 32];   // [m][k] ping-pong
  __shared__ unsigned short Bl[2][64 * 32];   // transposed [n][k] ping-pong
  const int tid = threadIdx.x, lane = tid & 31, wid = tid >> 5;
  const int bm = blockIdx.y * 64, bn = blockIdx.x * 64;
  const int wm = (wid & 3) * 16, wn = (wid >> 2) * 32;
  const int l15 = lane & 15, half = lane >> 4;
  const int fA = tid * 8, rA = fA >> 5, cA = fA & 31;   // A-tile chunk coords
  const int fB = tid * 8, rB = fB >> 6, nB = fB & 63;   // B-tile chunk coords
  v8f acc0 = {}, acc1 = {};

  u32x4 breg;
  async_copy_b128(&A[(bm + rA) * K + cA], &Al[0][fA]);          // A tile 0
  breg = *(const u32x4*)&B[rB * N + bn + nB];                   // B tile 0

  const int T = K / 32;
  for (int i = 0; i < T; ++i) {
    const int buf = i & 1;
    {  // publish current B tile (transposed) from registers
      const unsigned short* ts = (const unsigned short*)&breg;
#pragma unroll
      for (int j = 0; j < 8; ++j) Bl[buf][(nB + j) * 32 + rB] = ts[j];
    }
    if (i + 1 < T) {  // launch next tile while this one computes
      int kt2 = (i + 1) * 32;
      async_copy_b128(&A[(bm + rA) * K + kt2 + cA], &Al[buf ^ 1][fA]);
      breg = *(const u32x4*)&B[(kt2 + rB) * N + bn + nB];
      wait_async1();  // tile i landed; tile i+1 still in flight
    } else {
      wait_async0();
    }
    __syncthreads();
    Frag16 a, b0, b1;
    const unsigned short* ar = &Al[buf][(wm + l15) * 32];
    a.q[0] = *(const u32x4*)(ar + half * 16);
    a.q[1] = *(const u32x4*)(ar + half * 16 + 8);
    const unsigned short* b0p = &Bl[buf][(wn + l15) * 32 + half * 16];
    b0.q[0] = *(const u32x4*)(b0p);
    b0.q[1] = *(const u32x4*)(b0p + 8);
    const unsigned short* b1p = &Bl[buf][(wn + 16 + l15) * 32 + half * 16];
    b1.q[0] = *(const u32x4*)(b1p);
    b1.q[1] = *(const u32x4*)(b1p + 8);
    acc0 = __builtin_amdgcn_wmma_f32_16x16x32_bf16(false, a.v, false, b0.v,
                                                   (short)0, acc0, false, false);
    acc1 = __builtin_amdgcn_wmma_f32_16x16x32_bf16(false, a.v, false, b1.v,
                                                   (short)0, acc1, false, false);
    __syncthreads();  // fences ping-pong WAR for next iteration's writers
  }
  const int row0 = bm + wm + half * 8;
  const int c0 = bn + wn + l15, c1 = c0 + 16;
#pragma unroll
  for (int r = 0; r < 8; ++r) {
    int row = row0 + r;
    if constexpr (OUT_BF16) {
      Cb[row * N + c0] = f2bf(acc0[r]);
      Cb[row * N + c1] = f2bf(acc1[r]);
    } else {
      Cf[row * N + c0] = acc0[r] + (bias ? bias[c0] : 0.f);
      Cf[row * N + c1] = acc1[r] + (bias ? bias[c1] : 0.f);
    }
  }
}

// ------------------------------------------------- window means of q and k
__global__ void mean_qk(const unsigned short* __restrict__ qkv,
                        float* __restrict__ qm, float* __restrict__ km) {
  int bid = blockIdx.x;  // BATCH*HEADS*WS blocks, HD threads
  int t = bid & 255, h = (bid >> 8) & 7, b = bid >> 11;
  int dd = threadIdx.x;
  float sq = 0.f, sk = 0.f;
  for (int w = 0; w < NW; ++w) {
    int base = (b * NTOK + w * WS + t) * (3 * DIM) + h * HD + dd;
    sq += bf2f(qkv[base]);
    sk += bf2f(qkv[base + DIM]);
  }
  int o = ((b * HEADS + h) * WS + t) * HD + dd;
  qm[o] = sq * (1.f / NW);
  km[o] = sk * (1.f / NW);
}

// --------------------------------------- global-token bias per (b,h): [WS]
__global__ __launch_bounds__(256) void global_bias_k(
    const float* __restrict__ gp, const float* __restrict__ qm,
    const float* __restrict__ km, float* __restrict__ gbias) {
  __shared__ float gq[G * HD];
  int h = blockIdx.x & 7, b = blockIdx.x >> 3;
  int tid = threadIdx.x;
  const float* qmb = qm + (size_t)(b * HEADS + h) * WS * HD;
  const float* kmb = km + (size_t)(b * HEADS + h) * WS * HD;
  for (int i = 0; i < (G * HD) / 256; ++i) {
    int idx = tid + i * 256, g = idx >> 6, dd = idx & 63;
    float s = 0.f;
    for (int n = 0; n < WS; ++n) s += gp[(h * G + g) * WS + n] * qmb[n * HD + dd];
    gq[idx] = s;
  }
  __syncthreads();
  int m = tid;
  float s2 = 0.f;
  for (int g = 0; g < G; ++g) {
    float dot = 0.f;
#pragma unroll 8
    for (int dd = 0; dd < HD; ++dd) dot += gq[g * HD + dd] * kmb[m * HD + dd];
    s2 += dot;
  }
  gbias[(b * HEADS + h) * WS + m] = s2 * SCALE * (1.f / G);
}

// ---------------------------------------------- fused per-window attention
// One block per (b,h,w). Dynamic LDS = 256KB scores + 32KB q + 32KB k = 320KB
// (CDNA5 WGP LDS capacity). q buffer is reused for v^T before the AV matmul.
// q/k window tiles are fetched by the Tensor Data Mover (one descriptor each).
__global__ __launch_bounds__(256) void attn_window(
    const unsigned short* __restrict__ qkv, const float* __restrict__ lbias,
    const float* __restrict__ gbias, const float* __restrict__ gate,
    unsigned short* __restrict__ aout) {
  extern __shared__ char smem[];
  float* S = (float*)smem;                                     // [WS][WS] f32
  unsigned short* qs = (unsigned short*)(smem + WS * WS * 4);  // [WS][HD] bf16
  unsigned short* ks = qs + WS * HD;                           // [WS][HD] bf16

  const int blk = blockIdx.x;
  const int w = blk & (NW - 1), h = (blk >> 5) & 7, b = blk >> 8;
  const int tid = threadIdx.x, lane = tid & 31, wid = tid >> 5;
  const int l15 = lane & 15, half = lane >> 4;
  const int rowbase = b * NTOK + w * WS;
  const float sg = 1.f / (1.f + __expf(-gate[h]));
  const float* gb = gbias + (b * HEADS + h) * WS;
  const float* lb = lbias + h * WS * WS;

#ifdef HAVE_TDM
  if (wid == 0) {  // one wave issues both tile DMAs; TENSORcnt tracks them
    const unsigned short* gq = &qkv[(size_t)rowbase * (3 * DIM) + h * HD];
    tdm_load_2d_bf16(gq,       lds_off(qs), HD, WS, 3 * DIM);
    tdm_load_2d_bf16(gq + DIM, lds_off(ks), HD, WS, 3 * DIM);
    wait_tensor0();
  }
#else
  for (int i = 0; i < (WS * HD) / (256 * 8); ++i) {  // async b128 fallback
    int c = (tid + i * 256) * 8;
    int t = c >> 6, dd = c & 63;
    const unsigned short* gq = &qkv[(rowbase + t) * (3 * DIM) + h * HD + dd];
    async_copy_b128(gq, &qs[c]);
    async_copy_b128(gq + DIM, &ks[c]);
  }
  wait_async0();
#endif
  __syncthreads();

  // S = q k^T * scale + local_bias + sg * gbias   (256 tiles of 16x16, 8 waves)
  for (int tt = 0; tt < 32; ++tt) {
    int idx = wid * 32 + tt, ti = idx >> 4, tj = idx & 15;
    v8f acc = {};
#pragma unroll
    for (int kb = 0; kb < HD; kb += 32) {
      Frag16 a, bb;
      const unsigned short* ar = &qs[(ti * 16 + l15) * HD + kb];
      a.q[0] = *(const u32x4*)(ar + half * 16);
      a.q[1] = *(const u32x4*)(ar + half * 16 + 8);
      const unsigned short* br = &ks[(tj * 16 + l15) * HD + kb + half * 16];
      bb.q[0] = *(const u32x4*)(br);
      bb.q[1] = *(const u32x4*)(br + 8);
      acc = __builtin_amdgcn_wmma_f32_16x16x32_bf16(false, a.v, false, bb.v,
                                                    (short)0, acc, false, false);
    }
#pragma unroll
    for (int r = 0; r < 8; ++r) {
      int m = ti * 16 + r + half * 8, n = tj * 16 + l15;
      S[m * WS + n] = acc[r] * SCALE + lb[m * WS + n] + sg * gb[n];
    }
  }
  __syncthreads();

  // stage v^T[HD][WS] into (retired) q buffer (transpose -> synchronous path)
  for (int i = 0; i < (WS * HD) / 256; ++i) {
    int e = tid + i * 256, t = e >> 6, dd = e & 63;
    qs[dd * WS + t] = qkv[(rowbase + t) * (3 * DIM) + 2 * DIM + h * HD + dd];
  }

  // exact per-row top-k(179) threshold via radix binary search + softmax
  for (int rr = 0; rr < 32; ++rr) {
    int row = wid * 32 + rr;
    float vals[8]; unsigned keys[8];
#pragma unroll
    for (int j = 0; j < 8; ++j) {
      vals[j] = S[row * WS + j * 32 + lane];
      keys[j] = keyOf(vals[j]);
    }
    unsigned lo = 0u, hi = 0xFFFFFFFFu;  // invariant: count(>=lo) >= KKEEP
    while (lo < hi) {
      unsigned mid = lo + ((hi - lo) >> 1) + 1u;
      int c = 0;
#pragma unroll
      for (int j = 0; j < 8; ++j) c += (keys[j] >= mid) ? 1 : 0;
#pragma unroll
      for (int off = 16; off > 0; off >>= 1) c += __shfl_xor(c, off, 32);
      if (c >= KKEEP) lo = mid; else hi = mid - 1u;
    }
    float mx = -3.4e38f;
#pragma unroll
    for (int j = 0; j < 8; ++j) if (keys[j] >= lo) mx = fmaxf(mx, vals[j]);
#pragma unroll
    for (int off = 16; off > 0; off >>= 1) mx = fmaxf(mx, __shfl_xor(mx, off, 32));
    float p[8], sum = 0.f;
#pragma unroll
    for (int j = 0; j < 8; ++j) {
      p[j] = (keys[j] >= lo) ? __expf(vals[j] - mx) : 0.f;
      sum += p[j];
    }
#pragma unroll
    for (int off = 16; off > 0; off >>= 1) sum += __shfl_xor(sum, off, 32);
    float inv = 1.f / sum;
#pragma unroll
    for (int j = 0; j < 8; ++j) S[row * WS + j * 32 + lane] = p[j] * inv;
  }
  __syncthreads();

  // O = P @ v   (P fp32 in LDS -> bf16 frags on the fly; v^T in qs)
  for (int tt = 0; tt < 8; ++tt) {
    int idx = wid * 8 + tt, ti = idx >> 2, tj = idx & 3;
    v8f acc = {};
    for (int kb = 0; kb < WS; kb += 32) {
      Frag16 a, bb;
      const float* pr = &S[(ti * 16 + l15) * WS + kb];
#pragma unroll
      for (int j = 0; j < 8; ++j) {
        a.u[j]     = f2bf(pr[half * 16 + j]);
        a.u[8 + j] = f2bf(pr[half * 16 + 8 + j]);
      }
      const unsigned short* br = &qs[(tj * 16 + l15) * WS + kb + half * 16];
      bb.q[0] = *(const u32x4*)(br);
      bb.q[1] = *(const u32x4*)(br + 8);
      acc = __builtin_amdgcn_wmma_f32_16x16x32_bf16(false, a.v, false, bb.v,
                                                    (short)0, acc, false, false);
    }
#pragma unroll
    for (int r = 0; r < 8; ++r) {
      int t = ti * 16 + r + half * 8, dd = tj * 16 + l15;
      aout[(size_t)(rowbase + t) * DIM + h * HD + dd] = f2bf(acc[r]);
    }
  }
}

// ---------------------------------------------------------------- launcher
extern "C" void kernel_launch(void* const* d_in, const int* in_sizes, int n_in,
                              void* d_out, int out_size, void* d_ws,
                              size_t ws_size, hipStream_t stream) {
  const float* x     = (const float*)d_in[0];
  const float* wqkv  = (const float*)d_in[1];
  const float* wout  = (const float*)d_in[2];
  const float* bout  = (const float*)d_in[3];
  const float* lbias = (const float*)d_in[4];
  const float* gp    = (const float*)d_in[5];
  const float* gate  = (const float*)d_in[6];
  float* out = (float*)d_out;

  char* ws = (char*)d_ws;
  size_t off = 0;
  auto alloc = [&](size_t bytes) {
    char* p = ws + off;
    off += (bytes + 255) & ~size_t(255);
    return p;
  };
  unsigned short* xb    = (unsigned short*)alloc((size_t)BATCH * NTOK * DIM * 2);
  unsigned short* wqkvb = (unsigned short*)alloc((size_t)DIM * 3 * DIM * 2);
  unsigned short* woutb = (unsigned short*)alloc((size_t)DIM * DIM * 2);
  unsigned short* qkvb  = (unsigned short*)alloc((size_t)BATCH * NTOK * 3 * DIM * 2);
  float* qm    = (float*)alloc((size_t)BATCH * HEADS * WS * HD * 4);
  float* km    = (float*)alloc((size_t)BATCH * HEADS * WS * HD * 4);
  float* gbias = (float*)alloc((size_t)BATCH * HEADS * WS * 4);
  unsigned short* aoutb = (unsigned short*)alloc((size_t)BATCH * NTOK * DIM * 2);

  const int M = BATCH * NTOK;  // 32768
  cvt_f32_bf16<<<4096, 256, 0, stream>>>(x, xb, M * DIM);
  cvt_f32_bf16<<<512, 256, 0, stream>>>(wqkv, wqkvb, DIM * 3 * DIM);
  cvt_f32_bf16<<<256, 256, 0, stream>>>(wout, woutb, DIM * DIM);

  {  // QKV projection (bf16 out)
    dim3 g(3 * DIM / 64, M / 64);
    gemm_bf16<true><<<g, 256, 0, stream>>>(xb, wqkvb, nullptr, qkvb, nullptr,
                                           M, 3 * DIM, DIM);
  }
  mean_qk<<<BATCH * HEADS * WS, HD, 0, stream>>>(qkvb, qm, km);
  global_bias_k<<<BATCH * HEADS, 256, 0, stream>>>(gp, qm, km, gbias);

  attn_window<<<BATCH * HEADS * NW, 256, WS * WS * 4 + 2 * WS * HD * 2,
                stream>>>(qkvb, lbias, gbias, gate, aoutb);

  {  // output projection (+bias, fp32 out)
    dim3 g(DIM / 64, M / 64);
    gemm_bf16<false><<<g, 256, 0, stream>>>(aoutb, woutb, out, nullptr, bout,
                                            M, DIM, DIM);
  }
  (void)in_sizes; (void)n_in; (void)out_size; (void)ws_size;
}
